// DSAttention_858993459684
// MI455X (gfx1250) — matrix operand order repaired and checked by reading, not compiled, not run on hardware
//
#include <hip/hip_runtime.h>

typedef float    v2f  __attribute__((ext_vector_type(2)));
typedef float    v8f  __attribute__((ext_vector_type(8)));
typedef _Float16 v16h __attribute__((ext_vector_type(16)));
typedef unsigned v4u  __attribute__((ext_vector_type(4)));
typedef int      v4i  __attribute__((ext_vector_type(4)));
typedef int      v8i  __attribute__((ext_vector_type(8)));

#define B_  2
#define L_  2048
#define S_  2048
#define H_  8
#define E_  64
#define D_  64
#define BM  64     // query rows per block (16 per wave, 4 waves)
#define BN  32     // key cols per iteration
#define KS  68     // padded f32 row stride for K tiles = 64 data + 4 pad DWORDs (TDM pad)
#define VS  34     // padded f16 row stride for V^T / P tiles
#define NW  4
#define SCALE 0.125f   // 1/sqrt(64)

union Frag16 { v16h v; unsigned u[8]; };

__device__ __forceinline__ float rowmax16(float x) {
  x = fmaxf(x, __shfl_xor(x, 8, 32));
  x = fmaxf(x, __shfl_xor(x, 4, 32));
  x = fmaxf(x, __shfl_xor(x, 2, 32));
  x = fmaxf(x, __shfl_xor(x, 1, 32));
  return x;
}
__device__ __forceinline__ float rowsum16(float x) {
  x += __shfl_xor(x, 8, 32);
  x += __shfl_xor(x, 4, 32);
  x += __shfl_xor(x, 2, 32);
  x += __shfl_xor(x, 1, 32);
  return x;
}

__device__ __forceinline__ unsigned lds_off(const void* p) {
  // LDS aperture: flat addr low 32 bits are the wave-relative LDS byte offset.
  return (unsigned)(unsigned long long)p;
}

// TDM: DMA one 32x64 f32 K tile (row stride H_*E_ elems) into LDS with
// +4-DWORD padding every 64 DWORDs -> 68-float LDS rows (== KS).
__device__ __forceinline__ void tdm_load_k(unsigned lds_byte, const float* gptr) {
  unsigned long long ga = (unsigned long long)gptr;
  v4u g0;
  g0.x = 1u;                                            // count=1, no gather
  g0.y = lds_byte;                                      // lds_addr (bytes)
  g0.z = (unsigned)(ga & 0xffffffffu);                  // global_addr[31:0]
  g0.w = (unsigned)((ga >> 32) & 0x01ffffffu) | (2u << 30); // addr[56:32] | type=2
  v8i g1;
  g1[0] = (2 << 16) | (1 << 20) | (5 << 22) | (3 << 25); // data_size=4B, pad_en,
                                                         // pad_interval=64dw, pad_amount=4dw
  g1[1] = (int)(64u << 16);                              // tensor_dim0 = 64  (bits 79:48 lo)
  g1[2] = (int)(2048u << 16);                            // tensor_dim1 = 2048 (bits 111:80 lo)
  g1[3] = (int)(64u << 16);                              // tile_dim0 = 64   (bits 127:112)
  g1[4] = 32;                                            // tile_dim1 = 32
  g1[5] = (int)(H_ * E_);                                // tensor_dim0_stride = 512
  g1[6] = 0;
  g1[7] = 0;
  v4i z = {0, 0, 0, 0};
#if __clang_major__ >= 23
  v8i z8 = {0, 0, 0, 0, 0, 0, 0, 0};
  __builtin_amdgcn_tensor_load_to_lds(g0, g1, z, z, z8, 0);
#else
  __builtin_amdgcn_tensor_load_to_lds(g0, g1, z, z, 0);
#endif
}

// One 16(q-rows) x 32(k-cols) attention step for one branch.
__device__ __forceinline__ void attn_step(
    const float* __restrict__ sK, const _Float16* __restrict__ sV,
    _Float16* __restrict__ lPw, const v2f* qa,
    const int* kbA, const int* kbB,
    float taub, float d0, float d1,
    int lbase, int s0, int s1, int ln16, int hi,
    float* mrow, float* lrow, v8f* o)
{
  // ---- scores: Q(16x64) * K^T -> two 16x16 f32 tiles via V_WMMA_F32_16X16X4_F32
  v8f acc0 = {}, acc1 = {};
  #pragma unroll
  for (int kk = 0; kk < 16; ++kk) {
    v2f b0 = *(const v2f*)(sK + (ln16)      * KS + 4 * kk + 2 * hi);
    v2f b1 = *(const v2f*)(sK + (16 + ln16) * KS + 4 * kk + 2 * hi);
    acc0 = __builtin_amdgcn_wmma_f32_16x16x4_f32(false, qa[kk], false, b0,
                                                 (short)0, acc0, false, false);
    acc1 = __builtin_amdgcn_wmma_f32_16x16x4_f32(false, qa[kk], false, b1,
                                                 (short)0, acc1, false, false);
  }

  // ---- online softmax (row m = r + 8*hi lives across lanes 0-15 / 16-31)
  #pragma unroll
  for (int r = 0; r < 8; ++r) {
    const int l = lbase + r;
    float g0 = (s0 <= l) ? SCALE * (acc0[r] * taub + d0) : -1e30f;
    float g1 = (s1 <= l) ? SCALE * (acc1[r] * taub + d1) : -1e30f;
    float mx = rowmax16(fmaxf(g0, g1));
    float nm = fmaxf(mrow[r], mx);
    float a  = __expf(mrow[r] - nm);
    float e0 = __expf(g0 - nm);
    float e1 = __expf(g1 - nm);
    lrow[r] = lrow[r] * a + rowsum16(e0 + e1);
    mrow[r] = nm;
    #pragma unroll
    for (int t = 0; t < 4; ++t) o[t][r] *= a;
    // stash P in D-layout positions; LDS roundtrip performs the transpose
    lPw[(r + 8 * hi) * VS + ln16]      = (_Float16)e0;
    lPw[(r + 8 * hi) * VS + 16 + ln16] = (_Float16)e1;
  }

  // ---- P (16x32 f16) A-fragment read-back (same-wave LDS ordering)
  Frag16 pa;
  #pragma unroll
  for (int i = 0; i < 8; ++i)
    pa.u[i] = *(const unsigned*)(lPw + ln16 * VS + kbA[i]);

  // ---- O += P * V via V_WMMA_F32_16X16X32_F16 (4 d-tiles)
  #pragma unroll
  for (int t = 0; t < 4; ++t) {
    Frag16 vb;
    #pragma unroll
    for (int i = 0; i < 8; ++i)
      vb.u[i] = *(const unsigned*)(sV + (t * 16 + ln16) * VS + kbB[i]);
    o[t] = __builtin_amdgcn_wmma_f32_16x16x32_f16(false, pa.v, false, vb.v,
                                                  (short)0, o[t], false, false);
  }
}

__global__ __launch_bounds__(128)
void dsattn_wmma_kernel(const float* __restrict__ Q,  const float* __restrict__ Ke,
                        const float* __restrict__ Ve, const float* __restrict__ Kn,
                        const float* __restrict__ Vn, const float* __restrict__ tau,
                        const float* __restrict__ delta, const float* __restrict__ smask,
                        float* __restrict__ out)
{
  __shared__ float    sKe2[2][BN * KS];    // TDM ping-pong
  __shared__ float    sKn2[2][BN * KS];
  __shared__ _Float16 sVe[D_ * VS];        // transposed: [d][k], f16
  __shared__ _Float16 sVn[D_ * VS];
  __shared__ _Float16 sP [NW * 16 * VS];   // per-wave P transpose scratch

  const int tid  = threadIdx.x;
  const int wave = tid >> 5;
  const int lane = tid & 31;
  const int ln16 = lane & 15;
  const int hi   = lane >> 4;
  const int b    = blockIdx.z;
  const int h    = blockIdx.y;
  const int m0   = blockIdx.x * BM;

  const float taub = tau[b];

  // Q rows for this wave, held in 32-bit A-fragment layout
  const int qrow = m0 + wave * 16 + ln16;
  v2f qa[16];
  {
    const float* qp = Q + ((size_t)(b * L_ + qrow) * H_ + h) * E_;
    #pragma unroll
    for (int kk = 0; kk < 16; ++kk)
      qa[kk] = *(const v2f*)(qp + 4 * kk + 2 * hi);
  }

  // f16 fragment dword k-offsets (ISA A/B layouts for 16-bit 16x32 / 32x16)
  int kbA[8], kbB[8];
  #pragma unroll
  for (int i = 0; i < 8; ++i) {
    kbA[i] = ((i < 4) ? 2 * i : 2 * i + 8) + 8 * hi;
    kbB[i] = ((i < 4) ? 2 * i : 2 * i + 8) + 8 * hi;
  }

  float me[8], le[8], mn[8], ln_[8];
  v8f   oe[4], on[4];
  const v8f vzero = {};
  #pragma unroll
  for (int r = 0; r < 8; ++r) { me[r] = mn[r] = -1e30f; le[r] = ln_[r] = 0.f; }
  #pragma unroll
  for (int t = 0; t < 4; ++t) { oe[t] = vzero; on[t] = vzero; }

  const int  lmaxw = m0 + wave * 16 + 15;
  const int  lbase = m0 + wave * 16 + 8 * hi;
  _Float16*  lPw   = sP + wave * 16 * VS;
  const int  jend  = m0 + BM;

  // ---- TDM prologue: prefetch K tiles for j0 = 0 into buffer 0 ----
  if (wave == 0) {
    size_t g0off = ((size_t)(b * S_) * H_ + h) * E_;
    tdm_load_k(lds_off(&sKe2[0][0]), Ke + g0off);
    tdm_load_k(lds_off(&sKn2[0][0]), Kn + g0off);
  }

  for (int j0 = 0; j0 < jend; j0 += BN) {
    const int buf = (j0 >> 5) & 1;
    __syncthreads();                       // prior reads of V and K[buf^1] done

    // ---- cooperative stage: V tiles transposed to f16 (manual: TDM can't convert) ----
    #pragma unroll
    for (int it = 0; it < 16; ++it) {
      int i  = it * 128 + tid;             // 0..2047
      int sl = i >> 6;                     // key row 0..31
      int d  = i & 63;
      size_t g = ((size_t)(b * S_ + j0 + sl) * H_ + h) * D_ + d;
      sVe[d * VS + sl] = (_Float16)Ve[g];
      sVn[d * VS + sl] = (_Float16)Vn[g];
    }

    // ---- TDM: issue next prefetch, then drain current tile's 2 loads ----
    if (wave == 0) {
      if (j0 + BN < jend) {
        size_t gn = ((size_t)(b * S_ + j0 + BN) * H_ + h) * E_;
        tdm_load_k(lds_off(&sKe2[buf ^ 1][0]), Ke + gn);
        tdm_load_k(lds_off(&sKn2[buf ^ 1][0]), Kn + gn);
        __builtin_amdgcn_s_wait_tensorcnt(2);   // current 2 done, prefetch in flight
      } else {
        __builtin_amdgcn_s_wait_tensorcnt(0);
      }
    }
    __syncthreads();                       // publish K[buf] + V to all waves

    if (j0 > lmaxw) continue;              // block fully masked for this wave

    const int   s0 = j0 + ln16, s1 = j0 + 16 + ln16;
    const float d0 = delta[b * S_ + s0];
    const float d1 = delta[b * S_ + s1];

    attn_step(sKe2[buf], sVe, lPw, qa, kbA, kbB, taub, d0, d1,
              lbase, s0, s1, ln16, hi, me, le, oe);
    attn_step(sKn2[buf], sVn, lPw, qa, kbA, kbB, taub, d0, d1,
              lbase, s0, s1, ln16, hi, mn, ln_, on);
  }

  // ---- epilogue: blend branches, normalize, store ----
  float ile[8], iln[8], smr[8];
  #pragma unroll
  for (int r = 0; r < 8; ++r) {
    ile[r] = 1.f / le[r];
    iln[r] = 1.f / ln_[r];
    smr[r] = smask[b * L_ + lbase + r];
  }
  #pragma unroll
  for (int t = 0; t < 4; ++t) {
    #pragma unroll
    for (int r = 0; r < 8; ++r) {
      const int l = lbase + r;
      float v = smr[r] * (oe[t][r] * ile[r]) + (1.f - smr[r]) * (on[t][r] * iln[r]);
      out[((size_t)(b * L_ + l) * H_ + h) * D_ + t * 16 + ln16] = v;
    }
  }
}

extern "C" void kernel_launch(void* const* d_in, const int* in_sizes, int n_in,
                              void* d_out, int out_size, void* d_ws, size_t ws_size,
                              hipStream_t stream) {
  (void)in_sizes; (void)n_in; (void)out_size; (void)d_ws; (void)ws_size;
  dim3 grid(L_ / BM, H_, B_);
  dim3 block(128);
  hipLaunchKernelGGL(dsattn_wmma_kernel, grid, block, 0, stream,
                     (const float*)d_in[0], (const float*)d_in[1],
                     (const float*)d_in[2], (const float*)d_in[3],
                     (const float*)d_in[4], (const float*)d_in[5],
                     (const float*)d_in[6], (const float*)d_in[7],
                     (float*)d_out);
}
